// DAT_3307124818654
// MI455X (gfx1250) — matrix operand order repaired and checked by reading, not compiled, not run on hardware
//
#include <hip/hip_runtime.h>
#include <hip/hip_bf16.h>

// ---------------------------------------------------------------------------
// Model constants
// ---------------------------------------------------------------------------
#define S    1024
#define D    1024
#define H    16
#define HD   64
#define NL   6
#define DFF  4096
#define EPSF 1.1920929e-07f
#define SD   (S * D)

typedef __bf16 bf16;
typedef __attribute__((ext_vector_type(16))) __bf16 v16bf;
typedef __attribute__((ext_vector_type(8)))  __bf16 v8bf;
typedef __attribute__((ext_vector_type(8)))  float  v8f;

static __device__ __forceinline__ v16bf cat8(v8bf lo, v8bf hi) {
    return __builtin_shufflevector(lo, hi, 0,1,2,3,4,5,6,7,8,9,10,11,12,13,14,15);
}

// ---------------------------------------------------------------------------
// f32 -> bf16 conversion (weights, one time per launch)
// ---------------------------------------------------------------------------
__global__ void cvt_bf16(const float* __restrict__ w, bf16* __restrict__ wb, long long n) {
    long long i = (long long)blockIdx.x * 256 + threadIdx.x;
    if (i < n) wb[i] = (bf16)w[i];
}

// ---------------------------------------------------------------------------
// Row RMS norm: one block (256 thr = 8 waves) per row of `cols` floats
// ---------------------------------------------------------------------------
template <typename OUT>
__global__ __launch_bounds__(256)
void rms_rows(const float* __restrict__ in, OUT* __restrict__ out, int cols) {
    const int row = blockIdx.x;
    const float* p = in + (size_t)row * cols;
    float ss = 0.f;
    for (int c = threadIdx.x; c < cols; c += 256) { float t = p[c]; ss += t * t; }
    ss += __shfl_xor(ss, 16, 32);
    ss += __shfl_xor(ss,  8, 32);
    ss += __shfl_xor(ss,  4, 32);
    ss += __shfl_xor(ss,  2, 32);
    ss += __shfl_xor(ss,  1, 32);
    __shared__ float red[8];
    if ((threadIdx.x & 31) == 0) red[threadIdx.x >> 5] = ss;
    __syncthreads();
    float tot = red[0] + red[1] + red[2] + red[3] + red[4] + red[5] + red[6] + red[7];
    float r = rsqrtf(tot / (float)cols + EPSF);
    OUT* o = out + (size_t)row * cols;
    for (int c = threadIdx.x; c < cols; c += 256) o[c] = (OUT)(p[c] * r);
}

// ---------------------------------------------------------------------------
// GEMM: C[M,N] = A[M,K](bf16) @ W[N,K]^T(bf16)  (+ R[M,N])
// Register-blocked: one wave computes a 32x64 strip = 2 M-tiles x 4 N-tiles
// (8 C tiles, 8 WMMAs per K-step of 32 against 12 b128 loads -> each A
// operand reused 4x, each B operand reused 2x).
// A layout (16x32, 16-bit): lane m=lane%16, K pairs at {8*half, 16+8*half}.
// B layout (32x16, 16-bit): lane n=lane%16, K pairs at {8*half, 16+8*half}.
// Requires M % 32 == 0, N % 64 == 0.
// ---------------------------------------------------------------------------
__global__ __launch_bounds__(256)
void gemm_bf16_nt(const bf16* __restrict__ A, const bf16* __restrict__ W,
                  const float* R, float* C, int M, int N, int K) {
    const int lane = threadIdx.x & 31;
    const int wave = threadIdx.x >> 5;
    const int half = lane >> 4;
    const int mr   = lane & 15;
    const int jobsN = N >> 6;                 // 64-wide column strips
    const int job = blockIdx.x * 8 + wave;
    const int jm = job / jobsN;               // 32-row block
    const int jn = job % jobsN;               // 64-col block

    const bf16* arow0 = A + (size_t)(jm * 32 + mr) * K;       // M tile 0
    const bf16* arow1 = arow0 + (size_t)16 * K;               // M tile 1
    const bf16* brow  = W + (size_t)(jn * 64 + mr) * K;       // N tile 0 (tiles spaced 16*K)

    v8f acc[2][4] = {};
    for (int k0 = 0; k0 < K; k0 += 32) {
        __builtin_prefetch(arow0 + k0 + 512, 0, 3);
        __builtin_prefetch(arow1 + k0 + 512, 0, 3);
        v16bf a0 = cat8(*(const v8bf*)(arow0 + k0 + 8 * half),
                        *(const v8bf*)(arow0 + k0 + 16 + 8 * half));
        v16bf a1 = cat8(*(const v8bf*)(arow1 + k0 + 8 * half),
                        *(const v8bf*)(arow1 + k0 + 16 + 8 * half));
#pragma unroll
        for (int n = 0; n < 4; ++n) {
            const bf16* bp = brow + (size_t)n * 16 * K + k0;
            __builtin_prefetch(bp + 512, 0, 3);
            v16bf b = cat8(*(const v8bf*)(bp + 8 * half),
                           *(const v8bf*)(bp + 16 + 8 * half));
            acc[0][n] = __builtin_amdgcn_wmma_f32_16x16x32_bf16(false, a0, false, b,
                                                                (short)0, acc[0][n], false, false);
            acc[1][n] = __builtin_amdgcn_wmma_f32_16x16x32_bf16(false, a1, false, b,
                                                                (short)0, acc[1][n], false, false);
        }
    }
#pragma unroll
    for (int mt = 0; mt < 2; ++mt) {
#pragma unroll
        for (int n = 0; n < 4; ++n) {
#pragma unroll
            for (int r = 0; r < 8; ++r) {
                int m  = jm * 32 + mt * 16 + r + 8 * half;
                int nn = jn * 64 + n * 16 + mr;
                size_t idx = (size_t)m * N + nn;
                float v = acc[mt][n][r];
                if (R) v += R[idx];
                C[idx] = v;
            }
        }
    }
}

// ---------------------------------------------------------------------------
// Per-head RMS + RoPE, writes head-major bf16 [H][S][HD]. scale folded in.
// One wave per (s, h); lane handles channels (lane, lane+32).
// ---------------------------------------------------------------------------
__global__ __launch_bounds__(32)
void rope_rms(const float* __restrict__ t, bf16* __restrict__ outh, float scale) {
    const int s = blockIdx.x >> 4;
    const int h = blockIdx.x & 15;
    const int lane = threadIdx.x;
    const float* p = t + (size_t)s * D + h * HD;
    float a = p[lane], b = p[lane + 32];
    float ss = a * a + b * b;
    ss += __shfl_xor(ss, 16, 32);
    ss += __shfl_xor(ss,  8, 32);
    ss += __shfl_xor(ss,  4, 32);
    ss += __shfl_xor(ss,  2, 32);
    ss += __shfl_xor(ss,  1, 32);
    float r = rsqrtf(ss * (1.0f / (float)HD) + EPSF);
    a *= r; b *= r;
    float inv = __powf(10000.f, -((float)(2 * lane) / (float)HD));
    float fr  = (float)s * inv;
    float co = __cosf(fr), sn = __sinf(fr);
    float o1 = ( a * co + b * sn) * scale;
    float o2 = (-a * sn + b * co) * scale;
    bf16* o = outh + ((size_t)h * S + s) * HD;
    o[lane]      = (bf16)o1;
    o[lane + 32] = (bf16)o2;
}

// v [S][D] f32 -> vt [H][HD][S] bf16 (channel-major for P@V B operand)
__global__ void vtrans(const float* __restrict__ v, bf16* __restrict__ vt) {
    int idx = blockIdx.x * 256 + threadIdx.x;   // over H*HD*S
    int sI = idx & (S - 1);
    int hc = idx >> 10;
    int hh = hc >> 6, c = hc & 63;
    vt[idx] = (bf16)v[(size_t)sI * D + hh * HD + c];
}

// Markov-mask prefix sum: cpad[j+1] = #(levels[0..j]==0)
__global__ void cpad_kernel(const int* __restrict__ levels, int* __restrict__ cpad) {
    if (threadIdx.x == 0 && blockIdx.x == 0) {
        int acc = 0; cpad[0] = 0;
        for (int j = 0; j < S; ++j) { acc += (levels[j] == 0); cpad[j + 1] = acc; }
    }
}

// xl = l0*x + l1*x0
__global__ void mix_kernel(const float* __restrict__ x, const float* __restrict__ x0,
                           const float* __restrict__ lambdas, int layer,
                           float* __restrict__ xl) {
    int i = blockIdx.x * 256 + threadIdx.x;
    xl[i] = lambdas[2 * layer] * x[i] + lambdas[2 * layer + 1] * x0[i];
}

// value residual mix; layer 0 captures v1
__global__ void vmix_kernel(float* v, float* v1, const float* __restrict__ lamb,
                            int layer, int n) {
    int i = blockIdx.x * 256 + threadIdx.x;
    if (i >= n) return;
    if (layer == 0) { v1[i] = v[i]; }
    else { float l = lamb[layer]; v[i] = (1.f - l) * v[i] + l * v1[i]; }
}

// h -> relu(h)^2 as bf16
__global__ void relusq_kernel(const float* __restrict__ h, bf16* __restrict__ hb) {
    size_t i = (size_t)blockIdx.x * 256 + threadIdx.x;
    float v = h[i];
    v = v > 0.f ? v * v : 0.f;
    hb[i] = (bf16)v;
}

// ---------------------------------------------------------------------------
// Flash attention: 1 wave per (head, 16-query block), key blocks of 32.
// QK^T via WMMA, masked online softmax (shuffle reductions within the 16-lane
// half that owns a row), P transposed C->A layout through LDS, P@V via WMMA.
// ---------------------------------------------------------------------------
__global__ __launch_bounds__(32)
void attn_flash(const bf16* __restrict__ qh, const bf16* __restrict__ kh,
                const bf16* __restrict__ vt,
                const int* __restrict__ levels, const int* __restrict__ samp,
                const int* __restrict__ cpad, bf16* __restrict__ att) {
    __shared__ __align__(16) bf16 ldsp[16 * 32];
    const int lane = threadIdx.x & 31;
    const int half = lane >> 4;
    const int mr   = lane & 15;
    const int h  = blockIdx.x >> 6;     // S/16 == 64 query blocks per head
    const int qb = blockIdx.x & 63;

    // Q as two A tiles (K = 0..31 and 32..63)
    const bf16* qrow = qh + ((size_t)h * S + qb * 16 + mr) * HD;
    v16bf a0 = cat8(*(const v8bf*)(qrow + 8 * half),      *(const v8bf*)(qrow + 16 + 8 * half));
    v16bf a1 = cat8(*(const v8bf*)(qrow + 32 + 8 * half), *(const v8bf*)(qrow + 48 + 8 * half));

    float mrun[8], lrun[8], qc[8];
    int   qs[8];
#pragma unroll
    for (int r = 0; r < 8; ++r) {
        int qi = qb * 16 + r + 8 * half;
        mrun[r] = -1e30f; lrun[r] = 0.f;
        qs[r] = samp[qi];
        qc[r] = (float)cpad[qi];
    }
    v8f yac[4] = {};

    const int kmax = qb * 16 + 15;
    for (int k0 = 0; k0 <= kmax; k0 += 32) {
        // ---- scores: two 16x16 tiles covering keys k0..k0+31 ----
        v8f s0 = {}, s1 = {};
        {
            const bf16* kr0 = kh + ((size_t)h * S + k0 + mr) * HD;
            v16bf b00 = cat8(*(const v8bf*)(kr0 + 8 * half),      *(const v8bf*)(kr0 + 16 + 8 * half));
            v16bf b01 = cat8(*(const v8bf*)(kr0 + 32 + 8 * half), *(const v8bf*)(kr0 + 48 + 8 * half));
            s0 = __builtin_amdgcn_wmma_f32_16x16x32_bf16(false, a0, false, b00, (short)0, s0, false, false);
            s0 = __builtin_amdgcn_wmma_f32_16x16x32_bf16(false, a1, false, b01, (short)0, s0, false, false);
            const bf16* kr1 = kh + ((size_t)h * S + k0 + 16 + mr) * HD;
            v16bf b10 = cat8(*(const v8bf*)(kr1 + 8 * half),      *(const v8bf*)(kr1 + 16 + 8 * half));
            v16bf b11 = cat8(*(const v8bf*)(kr1 + 32 + 8 * half), *(const v8bf*)(kr1 + 48 + 8 * half));
            s1 = __builtin_amdgcn_wmma_f32_16x16x32_bf16(false, a0, false, b10, (short)0, s1, false, false);
            s1 = __builtin_amdgcn_wmma_f32_16x16x32_bf16(false, a1, false, b11, (short)0, s1, false, false);
        }
        // ---- mask + online softmax ----
        const int k0p = k0 + mr, k1p = k0 + 16 + mr;
        const int ks0 = samp[k0p],   ks1 = samp[k1p];
        const int kl0 = levels[k0p], kl1 = levels[k1p];
        const float cp0 = (float)cpad[k0p + 1], cp1 = (float)cpad[k1p + 1];
        float p0[8], p1[8];
#pragma unroll
        for (int r = 0; r < 8; ++r) {
            int qi = qb * 16 + r + 8 * half;
            bool keep0 = (qi >= k0p) && (qs[r] == ks0) && !((kl0 == 0) && (qc[r] - cp0 > 0.f));
            bool keep1 = (qi >= k1p) && (qs[r] == ks1) && !((kl1 == 0) && (qc[r] - cp1 > 0.f));
            p0[r] = keep0 ? s0[r] : -1e30f;
            p1[r] = keep1 ? s1[r] : -1e30f;
            float bm = fmaxf(p0[r], p1[r]);
            bm = fmaxf(bm, __shfl_xor(bm, 1, 32));
            bm = fmaxf(bm, __shfl_xor(bm, 2, 32));
            bm = fmaxf(bm, __shfl_xor(bm, 4, 32));
            bm = fmaxf(bm, __shfl_xor(bm, 8, 32));
            float newm = fmaxf(mrun[r], bm);
            float f = __expf(mrun[r] - newm);
            mrun[r] = newm;
            p0[r] = __expf(p0[r] - newm);
            p1[r] = __expf(p1[r] - newm);
            float ps = p0[r] + p1[r];
            ps += __shfl_xor(ps, 1, 32);
            ps += __shfl_xor(ps, 2, 32);
            ps += __shfl_xor(ps, 4, 32);
            ps += __shfl_xor(ps, 8, 32);
            lrun[r] = lrun[r] * f + ps;
#pragma unroll
            for (int t = 0; t < 4; ++t) yac[t][r] *= f;
        }
        // ---- transpose P (C layout -> A layout) through LDS ----
        __syncthreads();
#pragma unroll
        for (int r = 0; r < 8; ++r) {
            ldsp[(r + 8 * half) * 32 + mr]      = (bf16)p0[r];
            ldsp[(r + 8 * half) * 32 + 16 + mr] = (bf16)p1[r];
        }
        __syncthreads();
        v16bf pa = cat8(*(const v8bf*)(ldsp + mr * 32 + 8 * half),
                        *(const v8bf*)(ldsp + mr * 32 + 16 + 8 * half));
        // ---- y += P @ V  (4 channel tiles of 16) ----
#pragma unroll
        for (int t = 0; t < 4; ++t) {
            const bf16* vr = vt + ((size_t)(h * HD + 16 * t + mr)) * S + k0;
            v16bf b = cat8(*(const v8bf*)(vr + 8 * half), *(const v8bf*)(vr + 16 + 8 * half));
            yac[t] = __builtin_amdgcn_wmma_f32_16x16x32_bf16(false, pa, false, b,
                                                             (short)0, yac[t], false, false);
        }
    }
    // ---- finalize: att[S][D] bf16 ----
#pragma unroll
    for (int r = 0; r < 8; ++r) {
        float inv = 1.0f / lrun[r];
        int qi = qb * 16 + r + 8 * half;
        size_t base = (size_t)qi * D + h * HD;
        att[base + mr]      = (bf16)(yac[0][r] * inv);
        att[base + 16 + mr] = (bf16)(yac[1][r] * inv);
        att[base + 32 + mr] = (bf16)(yac[2][r] * inv);
        att[base + 48 + mr] = (bf16)(yac[3][r] * inv);
    }
}

// ---------------------------------------------------------------------------
// Host orchestration
// ---------------------------------------------------------------------------
extern "C" void kernel_launch(void* const* d_in, const int* in_sizes, int n_in,
                              void* d_out, int out_size, void* d_ws, size_t ws_size,
                              hipStream_t stream) {
    (void)in_sizes; (void)n_in; (void)out_size; (void)ws_size;
    const float* x_in    = (const float*)d_in[0];
    const float* Wq      = (const float*)d_in[1];
    const float* Wk      = (const float*)d_in[2];
    const float* Wv      = (const float*)d_in[3];
    const float* Wo      = (const float*)d_in[4];
    const float* lamb    = (const float*)d_in[5];
    const float* lambdas = (const float*)d_in[6];
    const float* Wfc     = (const float*)d_in[7];
    const float* Wp      = (const float*)d_in[8];
    const int*   levels  = (const int*)d_in[9];
    const int*   sidx    = (const int*)d_in[10];

    size_t off = 0;
    auto alloc = [&](size_t bytes) -> char* {
        char* p = (char*)d_ws + off;
        off = (off + bytes + 255) & ~(size_t)255;
        return p;
    };
    float* xb  = (float*)alloc(SD * 4);
    float* x0v = (float*)alloc(SD * 4);
    float* xl  = (float*)alloc(SD * 4);
    float* q   = (float*)alloc(SD * 4);
    float* k   = (float*)alloc(SD * 4);
    float* v   = (float*)alloc(SD * 4);
    float* v1  = (float*)alloc(SD * 4);
    float* h1  = (float*)alloc((size_t)S * DFF * 4);
    bf16* xn   = (bf16*)alloc(SD * 2);
    bf16* qhB  = (bf16*)alloc(SD * 2);
    bf16* khB  = (bf16*)alloc(SD * 2);
    bf16* vtB  = (bf16*)alloc(SD * 2 + 256);
    bf16* attB = (bf16*)alloc(SD * 2);
    bf16* hb   = (bf16*)alloc((size_t)S * DFF * 2);
    int*  cp   = (int*)alloc((S + 1) * 4);
    bf16* WqB  = (bf16*)alloc((size_t)NL * D * D * 2);
    bf16* WkB  = (bf16*)alloc((size_t)NL * D * D * 2);
    bf16* WvB  = (bf16*)alloc((size_t)NL * D * D * 2);
    bf16* WoB  = (bf16*)alloc((size_t)NL * D * D * 2);
    bf16* WfcB = (bf16*)alloc((size_t)NL * DFF * D * 2);
    bf16* WpB  = (bf16*)alloc((size_t)NL * D * DFF * 2);

    const long long nw1 = (long long)NL * D * D;     // 6M
    const long long nw2 = (long long)NL * DFF * D;   // 24M
    const int g1 = (int)((nw1 + 255) / 256), g2 = (int)((nw2 + 255) / 256);
    cvt_bf16<<<g1, 256, 0, stream>>>(Wq, WqB, nw1);
    cvt_bf16<<<g1, 256, 0, stream>>>(Wk, WkB, nw1);
    cvt_bf16<<<g1, 256, 0, stream>>>(Wv, WvB, nw1);
    cvt_bf16<<<g1, 256, 0, stream>>>(Wo, WoB, nw1);
    cvt_bf16<<<g2, 256, 0, stream>>>(Wfc, WfcB, nw2);
    cvt_bf16<<<g2, 256, 0, stream>>>(Wp, WpB, nw2);

    rms_rows<float><<<S, 256, 0, stream>>>(x_in, xb, D);
    hipMemcpyAsync(x0v, xb, (size_t)SD * 4, hipMemcpyDeviceToDevice, stream);
    cpad_kernel<<<1, 32, 0, stream>>>(levels, cp);

    const int ew = SD / 256;                       // elementwise grid over S*D
    const int gDD  = (S / 32) * (D / 64) / 8;      // GEMM grid (32x64 strips), N = D
    const int gDFF = (S / 32) * (DFF / 64) / 8;    // GEMM grid, N = DFF

    for (int i = 0; i < NL; ++i) {
        mix_kernel<<<ew, 256, 0, stream>>>(xb, x0v, lambdas, i, xl);
        rms_rows<bf16><<<S, 256, 0, stream>>>(xl, xn, D);

        gemm_bf16_nt<<<gDD, 256, 0, stream>>>(xn, WqB + (size_t)i * D * D, nullptr, q, S, D, D);
        gemm_bf16_nt<<<gDD, 256, 0, stream>>>(xn, WkB + (size_t)i * D * D, nullptr, k, S, D, D);
        gemm_bf16_nt<<<gDD, 256, 0, stream>>>(xn, WvB + (size_t)i * D * D, nullptr, v, S, D, D);

        vmix_kernel<<<ew, 256, 0, stream>>>(v, v1, lamb, i, SD);
        rope_rms<<<S * H, 32, 0, stream>>>(q, qhB, 0.125f);   // HD^-0.5 folded into q
        rope_rms<<<S * H, 32, 0, stream>>>(k, khB, 1.0f);
        vtrans<<<ew, 256, 0, stream>>>(v, vtB);

        attn_flash<<<H * (S / 16), 32, 0, stream>>>(qhB, khB, vtB, levels, sidx, cp, attB);

        gemm_bf16_nt<<<gDD, 256, 0, stream>>>(attB, WoB + (size_t)i * D * D, xl, xb, S, D, D);

        rms_rows<bf16><<<S, 256, 0, stream>>>(xb, xn, D);
        gemm_bf16_nt<<<gDFF, 256, 0, stream>>>(xn, WfcB + (size_t)i * DFF * D, nullptr, h1, S, DFF, D);
        relusq_kernel<<<(int)((size_t)S * DFF / 256), 256, 0, stream>>>(h1, hb);
        gemm_bf16_nt<<<gDD, 256, 0, stream>>>(hb, WpB + (size_t)i * D * DFF, xb, xb, S, D, DFF);
    }

    rms_rows<float><<<S, 256, 0, stream>>>(xb, (float*)d_out, D);
}
